// WormNetCalcium_79139067396318
// MI455X (gfx1250) — compile-verified, matrix operand order checked by reading, not run on hardware
//
#include <hip/hip_runtime.h>

#define NN   302
#define TT   2048
#define TEE  2080
#define KK   33
#define ODOR 16
#define DTC  0.2f
#define KPAD 320
#define MPAD 304
#define NT   (NN*TT)

typedef __attribute__((ext_vector_type(16))) _Float16 v16h;
typedef __attribute__((ext_vector_type(8)))  _Float16 v8h;
typedef __attribute__((ext_vector_type(8)))  float    v8f;

// ---------------- Kernel A: sensory encode -> sens_ext (N, TE) ----------------
__global__ void k_sensory(const float* __restrict__ W_enc, const float* __restrict__ b_enc,
                          const float* __restrict__ mask,  const float* __restrict__ odor,
                          float* __restrict__ sens) {
    int n = blockIdx.x;
    float w[ODOR];
#pragma unroll
    for (int o = 0; o < ODOR; ++o) w[o] = W_enc[n * ODOR + o];
    float b = b_enc[n], m = mask[n];
    for (int te = threadIdx.x; te < TEE; te += blockDim.x) {
        float acc = b;
#pragma unroll
        for (int o = 0; o < ODOR; ++o) acc += w[o] * odor[o * TEE + te];
        sens[n * TEE + te] = m * acc;
    }
}

// ---------------- Kernel C: pad W_chem / W_elec to f16 (MPAD x KPAD) ----------
__global__ void k_wcvt(const float* __restrict__ Wc, const float* __restrict__ We,
                       _Float16* __restrict__ WcH, _Float16* __restrict__ WeH) {
    int idx = blockIdx.x * blockDim.x + threadIdx.x;
    if (idx >= MPAD * KPAD) return;
    int m = idx / KPAD, k = idx % KPAD;
    float c = 0.f, e = 0.f;
    if (m < NN && k < NN) { c = Wc[m * NN + k]; e = We[m * NN + k]; }
    WcH[idx] = (_Float16)c;
    WeH[idx] = (_Float16)e;
}

// ---------------- zero the K-pad columns of the transposed f16 activations ----
__global__ void k_padT(_Float16* __restrict__ actT, _Float16* __restrict__ sampT) {
    int idx = blockIdx.x * blockDim.x + threadIdx.x;
    const int PW = KPAD - NN; // 18
    if (idx >= TT * PW) return;
    int t = idx / PW, k = NN + idx % PW;
    actT[t * KPAD + k]  = (_Float16)0.f;
    sampT[t * KPAD + k] = (_Float16)0.f;
}

// ------- Kernel B: depthwise convs + reparam sample + relu/softplus; emits ----
// ------- f16 transposed act/sample for the WMMA GEMM. One block per neuron ----
__global__ void k_front(const float* __restrict__ fraw, const float* __restrict__ sens,
                        const float* __restrict__ eps,
                        const float* __restrict__ W_mu_f, const float* __restrict__ W_mu_s,
                        const float* __restrict__ b_mu,
                        const float* __restrict__ W_lv_f, const float* __restrict__ W_lv_s,
                        const float* __restrict__ b_lv,
                        const float* __restrict__ w_aff, const float* __restrict__ b_aff,
                        float* __restrict__ o_mu, float* __restrict__ o_lv,
                        float* __restrict__ o_sv, float* __restrict__ o_ca,
                        float* __restrict__ o_sens,
                        _Float16* __restrict__ actT, _Float16* __restrict__ sampT) {
    int n = blockIdx.x;
    __shared__ float wmf[KK], wms[KK], wlf[KK], wls[KK];
    if (threadIdx.x < KK) {
        wmf[threadIdx.x] = W_mu_f[n * KK + threadIdx.x];
        wms[threadIdx.x] = W_mu_s[n * KK + threadIdx.x];
        wlf[threadIdx.x] = W_lv_f[n * KK + threadIdx.x];
        wls[threadIdx.x] = W_lv_s[n * KK + threadIdx.x];
    }
    __syncthreads();
    float bmu = b_mu[n], blv = b_lv[n], wa = w_aff[n], ba = b_aff[n];
    const float* fr = fraw + (size_t)n * TEE;
    const float* se = sens + (size_t)n * TEE;
    for (int t = threadIdx.x; t < TT; t += blockDim.x) {
        float mu = bmu, lv = blv;
#pragma unroll
        for (int k = 0; k < KK; ++k) {
            float f = fr[t + k];
            float s = se[t + k];
            mu = fmaf(f, wmf[k], fmaf(s, wms[k], mu));
            lv = fmaf(f, wlf[k], fmaf(s, wls[k], lv));
        }
        float sv = mu + expf(0.5f * lv) * eps[n * TT + t];
        float a  = fmaxf(sv, 0.f);
        float x  = wa * sv + ba;
        float ca = (x > 20.f) ? x : log1pf(expf(x));
        size_t o = (size_t)n * TT + t;
        o_mu[o] = mu;  o_lv[o] = lv;  o_sv[o] = sv;  o_ca[o] = ca;
        o_sens[o] = se[t + 16];                 // crop = (TE - T)/2 = 16
        actT [(size_t)t * KPAD + n] = (_Float16)a;
        sampT[(size_t)t * KPAD + n] = (_Float16)sv;
    }
}

// ---------------- WMMA fragment load: A row-major (row, Kpad) -----------------
// lane l: M(or N)=l%16 ; elems 0..7 at kb+8*(l>=16), elems 8..15 at kb+16+8*(l>=16)
__device__ __forceinline__ v16h load_frag(const _Float16* p) {
    v8h lo = *(const v8h*)(p);
    v8h hi = *(const v8h*)(p + 16);
    v16h r;
#pragma unroll
    for (int i = 0; i < 8; ++i) { r[i] = lo[i]; r[i + 8] = hi[i]; }
    return r;
}

// ---- Kernel D: recurrent_in = Wc@act + We@samp (+sens) ; voltage update ------
// one wave32 per 16x16 output tile; K = 320 in 10 chunks of 32
__global__ __launch_bounds__(32) void k_gemm(
        const _Float16* __restrict__ WcH, const _Float16* __restrict__ WeH,
        const _Float16* __restrict__ actT, const _Float16* __restrict__ sampT,
        const float* __restrict__ bias, const float* __restrict__ tau,
        const float* __restrict__ sens_in, const float* __restrict__ sample_v,
        float* __restrict__ rec_out, float* __restrict__ volt_out) {
    int tb   = blockIdx.x << 4;       // T tile base
    int mb   = blockIdx.y << 4;       // M tile base
    int lane = threadIdx.x;
    int hi   = lane >> 4;             // 0 | 1
    int lr   = lane & 15;
    const _Float16* wc = WcH  + (size_t)(mb + lr) * KPAD + hi * 8;
    const _Float16* we = WeH  + (size_t)(mb + lr) * KPAD + hi * 8;
    const _Float16* at = actT + (size_t)(tb + lr) * KPAD + hi * 8;
    const _Float16* st = sampT+ (size_t)(tb + lr) * KPAD + hi * 8;

    v8f acc = {};
    for (int kb = 0; kb < KPAD; kb += 32) {
        v16h aC = load_frag(wc + kb);
        v16h bA = load_frag(at + kb);
        acc = __builtin_amdgcn_wmma_f32_16x16x32_f16(false, aC, false, bA,
                                                     (short)0, acc, false, false);
        v16h aE = load_frag(we + kb);
        v16h bS = load_frag(st + kb);
        acc = __builtin_amdgcn_wmma_f32_16x16x32_f16(false, aE, false, bS,
                                                     (short)0, acc, false, false);
    }

    int t = tb + lr;
#pragma unroll
    for (int v = 0; v < 8; ++v) {
        int m = mb + v + (hi << 3);   // C/D layout: VGPR v, lane-half -> row
        if (m < NN) {
            size_t o = (size_t)m * TT + t;
            float rec = acc[v] + sens_in[o];
            rec_out[o] = rec;
            float sv = sample_v[o];
            volt_out[o] = sv + (DTC / tau[m]) * (-sv + rec + bias[m]);
        }
    }
}

// ---- Kernel E: calcium IIR scan + fluorescence --------------------------------
// jnp.convolve(ca, exp(-t*alpha))[:T] + init*kern  ==  mc[t]=ca[t]+r*mc[t-1],
// seeded mc[0]=ca[0]+init, with r=exp(-alpha)
__global__ void k_scan(const float* __restrict__ ca, const float* __restrict__ ffull,
                       const float* __restrict__ scale, const float* __restrict__ shift,
                       const float* __restrict__ ctau,
                       float* __restrict__ mc_out, float* __restrict__ fl_out) {
    int n = blockIdx.x * blockDim.x + threadIdx.x;
    if (n >= NN) return;
    float sc = scale[n], sh = shift[n];
    float r  = expf(-DTC / ctau[n]);
    float init = (ffull[(size_t)n * TT] - sh) / sc;
    const float* c = ca + (size_t)n * TT;
    float m = c[0] + init;
    mc_out[(size_t)n * TT] = m;
    fl_out[(size_t)n * TT] = sc * m + sh;
    for (int t = 1; t < TT; ++t) {
        m = c[t] + r * m;
        size_t o = (size_t)n * TT + t;
        mc_out[o] = m;
        fl_out[o] = sc * m + sh;
    }
}

extern "C" void kernel_launch(void* const* d_in, const int* in_sizes, int n_in,
                              void* d_out, int out_size, void* d_ws, size_t ws_size,
                              hipStream_t stream) {
    (void)in_sizes; (void)n_in; (void)out_size; (void)ws_size;
    const float* fraw   = (const float*)d_in[0];
    const float* ffull  = (const float*)d_in[1];
    const float* odor   = (const float*)d_in[2];
    const float* eps    = (const float*)d_in[3];
    const float* W_enc  = (const float*)d_in[4];
    const float* b_enc  = (const float*)d_in[5];
    const float* mask   = (const float*)d_in[6];
    const float* W_mu_f = (const float*)d_in[7];
    const float* W_mu_s = (const float*)d_in[8];
    const float* b_mu   = (const float*)d_in[9];
    const float* W_lv_f = (const float*)d_in[10];
    const float* W_lv_s = (const float*)d_in[11];
    const float* b_lv   = (const float*)d_in[12];
    const float* W_chem = (const float*)d_in[13];
    const float* W_elec = (const float*)d_in[14];
    const float* bias   = (const float*)d_in[15];
    const float* tau    = (const float*)d_in[16];
    const float* w_aff  = (const float*)d_in[17];
    const float* b_aff  = (const float*)d_in[18];
    const float* scale  = (const float*)d_in[19];
    const float* shift  = (const float*)d_in[20];
    const float* ctau   = (const float*)d_in[21];

    float* out    = (float*)d_out;
    float* o_volt = out + 0 * (size_t)NT;  // mu_neuron_voltage_prob
    float* o_fl   = out + 1 * (size_t)NT;  // mu_fluorescence_trace_prob
    float* o_mu   = out + 2 * (size_t)NT;  // mu_v
    float* o_lv   = out + 3 * (size_t)NT;  // logvar_v
    float* o_sv   = out + 4 * (size_t)NT;  // sample_v
    float* o_ca   = out + 5 * (size_t)NT;  // calcium_activation
    float* o_mc   = out + 6 * (size_t)NT;  // mu_calcium_prob (1,N,T)
    float* o_rec  = out + 7 * (size_t)NT;  // recurrent_in
    float* o_sens = out + 8 * (size_t)NT;  // sensory_input

    char* ws = (char*)d_ws;
    size_t off = 0;
    float* sens_ext = (float*)(ws + off);
    off += (size_t)NN * TEE * sizeof(float);  off = (off + 255) & ~(size_t)255;
    _Float16* actT  = (_Float16*)(ws + off);
    off += (size_t)TT * KPAD * sizeof(_Float16); off = (off + 255) & ~(size_t)255;
    _Float16* sampT = (_Float16*)(ws + off);
    off += (size_t)TT * KPAD * sizeof(_Float16); off = (off + 255) & ~(size_t)255;
    _Float16* WcH   = (_Float16*)(ws + off);
    off += (size_t)MPAD * KPAD * sizeof(_Float16); off = (off + 255) & ~(size_t)255;
    _Float16* WeH   = (_Float16*)(ws + off);

    k_sensory<<<NN, 256, 0, stream>>>(W_enc, b_enc, mask, odor, sens_ext);
    k_wcvt<<<(MPAD * KPAD + 255) / 256, 256, 0, stream>>>(W_chem, W_elec, WcH, WeH);
    k_padT<<<(TT * (KPAD - NN) + 255) / 256, 256, 0, stream>>>(actT, sampT);
    k_front<<<NN, 256, 0, stream>>>(fraw, sens_ext, eps, W_mu_f, W_mu_s, b_mu,
                                    W_lv_f, W_lv_s, b_lv, w_aff, b_aff,
                                    o_mu, o_lv, o_sv, o_ca, o_sens, actT, sampT);
    k_gemm<<<dim3(TT / 16, MPAD / 16), 32, 0, stream>>>(WcH, WeH, actT, sampT,
                                    bias, tau, o_sens, o_sv, o_rec, o_volt);
    k_scan<<<(NN + 63) / 64, 64, 0, stream>>>(o_ca, ffull, scale, shift, ctau,
                                              o_mc, o_fl);
}
